// MultiHeadAttention_87033217286362
// MI455X (gfx1250) — matrix operand order repaired and checked by reading, not compiled, not run on hardware
//
#include <hip/hip_runtime.h>
#include <hip/hip_bf16.h>

typedef __attribute__((ext_vector_type(16))) _Float16 v16h;
typedef __attribute__((ext_vector_type(8)))  float    v8f;
typedef __attribute__((ext_vector_type(4)))  int      v4i;

#define DIM   1024
#define NHEAD 16
#define HD    64
#define SEQ   2048
#define BATCH 2
#define BN    (BATCH * SEQ)

#if __has_builtin(__builtin_amdgcn_global_load_async_to_lds_b128)
#define HAVE_ASYNC 1
#else
#define HAVE_ASYNC 0
#endif

static __device__ __forceinline__ v8f wmma_f16(v16h a, v16h b, v8f c) {
  return __builtin_amdgcn_wmma_f32_16x16x32_f16(false, a, false, b, (short)0, c,
                                                false, false);
}

// A fragment 16x32 (MxK), row-major source, row stride ld (halves).
static __device__ __forceinline__ v16h load_a16x32(const _Float16* base, int row0,
                                                   int col0, int ld) {
  int lane = threadIdx.x & 31;
  int m = lane & 15, g = lane >> 4;
  const _Float16* p = base + (size_t)(row0 + m) * ld + col0;
  v16h a;
#pragma unroll
  for (int e = 0; e < 8; ++e) {
    a[e]     = p[g * 16 + e];
    a[8 + e] = p[g * 16 + 8 + e];
  }
  return a;
}

// B fragment 32x16 (KxN). Source column-major: colmaj[(n)*ld + k].
static __device__ __forceinline__ v16h load_b32x16(const _Float16* colmaj, int n0,
                                                   int k0, int ld) {
  int lane = threadIdx.x & 31;
  int n = lane & 15, g = lane >> 4;
  const _Float16* p = colmaj + (size_t)(n0 + n) * ld + k0 + g * 16;
  v16h b;
#pragma unroll
  for (int e = 0; e < 16; ++e) b[e] = p[e];
  return b;
}

// 16-byte async global->LDS copy (ASYNCcnt-tracked); sync fallback.
static __device__ __forceinline__ void async_copy16(const _Float16* g, _Float16* l) {
#if HAVE_ASYNC
  __builtin_amdgcn_global_load_async_to_lds_b128(
      (__attribute__((address_space(1))) v4i*)(void*)g,
      (__attribute__((address_space(3))) v4i*)(void*)l, 0, 0);
#else
  *(int4*)l = *(const int4*)g;
#endif
}

static __device__ __forceinline__ void wait_async() {
#if HAVE_ASYNC
#if __has_builtin(__builtin_amdgcn_s_wait_asynccnt)
  __builtin_amdgcn_s_wait_asynccnt(0);
#else
  asm volatile("s_wait_asynccnt 0x0" ::: "memory");
#endif
#endif
}

// ---------------- pack kernels ----------------

__global__ void k_cvt_f16(const float* __restrict__ in, _Float16* __restrict__ out,
                          int n) {
  int i = blockIdx.x * 256 + threadIdx.x;
  if (i < n) out[i] = (_Float16)in[i];
}

// in[k*cols + n] (row-major K x N) -> out[n*rows + k] (column-major f16)
__global__ void k_transpose_cvt(const float* __restrict__ in,
                                _Float16* __restrict__ out, int rows, int cols) {
  int i = blockIdx.x * 256 + threadIdx.x;
  if (i < rows * cols) {
    int k = i / cols, n = i % cols;
    out[(size_t)n * rows + k] = (_Float16)in[i];
  }
}

// ---------------- QKV projection: 64x64 per wave, 16 WMMAs per k-step --------
__global__ void __launch_bounds__(32) k_qkv(const _Float16* __restrict__ xh,
                                            const _Float16* __restrict__ Wt,
                                            const float* __restrict__ bias,
                                            _Float16* __restrict__ Q,
                                            _Float16* __restrict__ K,
                                            _Float16* __restrict__ V) {
  int col0 = blockIdx.x * 64;  // feature in [0, 3*DIM)
  int row0 = blockIdx.y * 64;  // token in [0, BN)
  v8f acc[4][4] = {};
  for (int k0 = 0; k0 < DIM; k0 += 32) {
    v16h a0 = load_a16x32(xh, row0 +  0, k0, DIM);
    v16h a1 = load_a16x32(xh, row0 + 16, k0, DIM);
    v16h a2 = load_a16x32(xh, row0 + 32, k0, DIM);
    v16h a3 = load_a16x32(xh, row0 + 48, k0, DIM);
#pragma unroll
    for (int jj = 0; jj < 4; ++jj) {
      v16h b = load_b32x16(Wt, col0 + jj * 16, k0, DIM);
      acc[0][jj] = wmma_f16(a0, b, acc[0][jj]);
      acc[1][jj] = wmma_f16(a1, b, acc[1][jj]);
      acc[2][jj] = wmma_f16(a2, b, acc[2][jj]);
      acc[3][jj] = wmma_f16(a3, b, acc[3][jj]);
    }
  }
  int lane = threadIdx.x & 31;
  int n = lane & 15, g = lane >> 4;
#pragma unroll
  for (int jj = 0; jj < 4; ++jj) {
    int f  = col0 + jj * 16 + n;
    float bv = bias[f];
    int s = f >> 10;         // which of q/k/v (DIM = 1024)
    int c = f & (DIM - 1);
    int h = c >> 6, d = c & (HD - 1);
#pragma unroll
    for (int ii = 0; ii < 4; ++ii) {
#pragma unroll
      for (int r = 0; r < 8; ++r) {
        int t    = row0 + ii * 16 + 8 * g + r;
        int bidx = t >> 11, nn = t & (SEQ - 1);
        _Float16 hv = (_Float16)(acc[ii][jj][r] + bv);
        size_t bh = (size_t)bidx * NHEAD + h;
        if (s == 0)      Q[(bh * SEQ + nn) * HD + d] = hv;
        else if (s == 1) K[(bh * SEQ + nn) * HD + d] = hv;
        else             V[(bh * HD + d) * SEQ + nn] = hv;
      }
    }
  }
}

// ---------------- attention: 4 waves/block, K/V staged in LDS via async copy -
// Stage one 32-token chunk: K rows j..j+31 ([32][HD]) and V cols j..j+31
// ([HD][32]). 256 16B chunks each; 128 threads -> 2+2 copies per thread.
static __device__ __forceinline__ void stage_kv(const _Float16* Kb,
                                                const _Float16* Vb, int j,
                                                _Float16* kd, _Float16* vd,
                                                int tid) {
#pragma unroll
  for (int c0 = 0; c0 < 2; ++c0) {
    int c = tid + c0 * 128;
    async_copy16(Kb + (size_t)(j + (c >> 3)) * HD + (c & 7) * 8, kd + c * 8);
    async_copy16(Vb + (size_t)(c >> 2) * SEQ + j + (c & 3) * 8, vd + c * 8);
  }
}

__global__ void __launch_bounds__(128) k_attn(const _Float16* __restrict__ Q,
                                              const _Float16* __restrict__ K,
                                              const _Float16* __restrict__ V,
                                              _Float16* __restrict__ O) {
  __shared__ __align__(16) _Float16 Ksh[2][32 * HD];  // [tok][d]
  __shared__ __align__(16) _Float16 Vsh[2][HD * 32];  // [d][tok]
  __shared__ __align__(16) _Float16 Pt[4][16 * 32];   // per-wave P tile

  int tid = threadIdx.x;
  int w   = tid >> 5;
  int bh     = blockIdx.x >> 5;  // b*NHEAD + h   (SEQ/64 = 32 tiles per bh)
  int tile64 = blockIdx.x & 31;
  int bidx = bh >> 4, h = bh & 15;
  const _Float16* Qb = Q + (size_t)bh * SEQ * HD;
  const _Float16* Kb = K + (size_t)bh * SEQ * HD;
  const _Float16* Vb = V + (size_t)bh * HD * SEQ;
  int n0 = tile64 * 64 + w * 16;  // this wave's 16 query rows

  v16h aq0 = load_a16x32(Qb, n0, 0, HD);
  v16h aq1 = load_a16x32(Qb, n0, 32, HD);

  v8f o0 = {}, o1 = {}, o2 = {}, o3 = {};
  float mreg[8], lreg[8];
#pragma unroll
  for (int r = 0; r < 8; ++r) { mreg[r] = -1e30f; lreg[r] = 0.0f; }

  const float scale = 0.125f;  // hd^-0.5
  int lane = tid & 31;
  int nc = lane & 15, g = lane >> 4;
  _Float16* Pw = Pt[w];

  stage_kv(Kb, Vb, 0, Ksh[0], Vsh[0], tid);

  for (int j = 0; j < SEQ; j += 32) {
    int cur = (j >> 5) & 1;
    wait_async();       // my chunk-j async copies done
    __syncthreads();    // everyone's done; prev buffer free for restage
    if (j + 32 < SEQ) stage_kv(Kb, Vb, j + 32, Ksh[cur ^ 1], Vsh[cur ^ 1], tid);

    const _Float16* Kc = Ksh[cur];
    const _Float16* Vc = Vsh[cur];

    v8f s0 = {}, s1 = {};
    s0 = wmma_f16(aq0, load_b32x16(Kc, 0, 0, HD), s0);
    s0 = wmma_f16(aq1, load_b32x16(Kc, 0, 32, HD), s0);
    s1 = wmma_f16(aq0, load_b32x16(Kc, 16, 0, HD), s1);
    s1 = wmma_f16(aq1, load_b32x16(Kc, 16, 32, HD), s1);

    float p0[8], p1[8];
#pragma unroll
    for (int r = 0; r < 8; ++r) {
      float a0 = s0[r] * scale, a1 = s1[r] * scale;
      float mx = fmaxf(a0, a1);
      mx = fmaxf(mx, __shfl_xor(mx, 1, 32));
      mx = fmaxf(mx, __shfl_xor(mx, 2, 32));
      mx = fmaxf(mx, __shfl_xor(mx, 4, 32));
      mx = fmaxf(mx, __shfl_xor(mx, 8, 32));
      float mn = fmaxf(mreg[r], mx);
      float fc = __expf(mreg[r] - mn);
      mreg[r] = mn;
      p0[r] = __expf(a0 - mn);
      p1[r] = __expf(a1 - mn);
      float rs = p0[r] + p1[r];
      rs += __shfl_xor(rs, 1, 32);
      rs += __shfl_xor(rs, 2, 32);
      rs += __shfl_xor(rs, 4, 32);
      rs += __shfl_xor(rs, 8, 32);
      lreg[r] = lreg[r] * fc + rs;
      o0[r] *= fc; o1[r] *= fc; o2[r] *= fc; o3[r] *= fc;
    }

    // C-layout -> per-wave LDS 16x32 row-major (wave-local; DScnt in-order)
#pragma unroll
    for (int r = 0; r < 8; ++r) {
      Pw[(8 * g + r) * 32 + nc]      = (_Float16)p0[r];
      Pw[(8 * g + r) * 32 + 16 + nc] = (_Float16)p1[r];
    }
    v16h ap = load_a16x32(Pw, 0, 0, 32);
    o0 = wmma_f16(ap, load_b32x16(Vc, 0, 0, 32), o0);
    o1 = wmma_f16(ap, load_b32x16(Vc, 16, 0, 32), o1);
    o2 = wmma_f16(ap, load_b32x16(Vc, 32, 0, 32), o2);
    o3 = wmma_f16(ap, load_b32x16(Vc, 48, 0, 32), o3);
  }

#pragma unroll
  for (int r = 0; r < 8; ++r) {
    float inv = 1.0f / lreg[r];
    int t = n0 + 8 * g + r;
    size_t base = ((size_t)bidx * SEQ + t) * DIM + (size_t)h * HD;
    O[base + 0  + nc] = (_Float16)(o0[r] * inv);
    O[base + 16 + nc] = (_Float16)(o1[r] * inv);
    O[base + 32 + nc] = (_Float16)(o2[r] * inv);
    O[base + 48 + nc] = (_Float16)(o3[r] * inv);
  }
}

// ---------------- output projection: 64x64 per wave, f32 out ----------------
__global__ void __launch_bounds__(32) k_proj(const _Float16* __restrict__ Oh,
                                             const _Float16* __restrict__ Wt,
                                             const float* __restrict__ bias,
                                             float* __restrict__ out) {
  int col0 = blockIdx.x * 64;
  int row0 = blockIdx.y * 64;
  v8f acc[4][4] = {};
  for (int k0 = 0; k0 < DIM; k0 += 32) {
    v16h a0 = load_a16x32(Oh, row0 +  0, k0, DIM);
    v16h a1 = load_a16x32(Oh, row0 + 16, k0, DIM);
    v16h a2 = load_a16x32(Oh, row0 + 32, k0, DIM);
    v16h a3 = load_a16x32(Oh, row0 + 48, k0, DIM);
#pragma unroll
    for (int jj = 0; jj < 4; ++jj) {
      v16h b = load_b32x16(Wt, col0 + jj * 16, k0, DIM);
      acc[0][jj] = wmma_f16(a0, b, acc[0][jj]);
      acc[1][jj] = wmma_f16(a1, b, acc[1][jj]);
      acc[2][jj] = wmma_f16(a2, b, acc[2][jj]);
      acc[3][jj] = wmma_f16(a3, b, acc[3][jj]);
    }
  }
  int lane = threadIdx.x & 31;
  int n = lane & 15, g = lane >> 4;
#pragma unroll
  for (int jj = 0; jj < 4; ++jj) {
    float bv = bias[col0 + jj * 16 + n];
#pragma unroll
    for (int ii = 0; ii < 4; ++ii) {
#pragma unroll
      for (int r = 0; r < 8; ++r) {
        out[(size_t)(row0 + ii * 16 + 8 * g + r) * DIM + col0 + jj * 16 + n] =
            acc[ii][jj][r] + bv;
      }
    }
  }
}

extern "C" void kernel_launch(void* const* d_in, const int* in_sizes, int n_in,
                              void* d_out, int out_size, void* d_ws, size_t ws_size,
                              hipStream_t stream) {
  const float* x      = (const float*)d_in[0];
  const float* W_qkv  = (const float*)d_in[1];
  const float* b_qkv  = (const float*)d_in[2];
  const float* W_proj = (const float*)d_in[3];
  const float* b_proj = (const float*)d_in[4];

  char* w = (char*)d_ws;
  _Float16* xh     = (_Float16*)w; w += (size_t)BN * DIM * 2;
  _Float16* Wqkvt  = (_Float16*)w; w += (size_t)3 * DIM * DIM * 2;
  _Float16* Wprojt = (_Float16*)w; w += (size_t)DIM * DIM * 2;
  _Float16* Qh     = (_Float16*)w; w += (size_t)BATCH * NHEAD * SEQ * HD * 2;
  _Float16* Kh     = (_Float16*)w; w += (size_t)BATCH * NHEAD * SEQ * HD * 2;
  _Float16* Vth    = (_Float16*)w; w += (size_t)BATCH * NHEAD * HD * SEQ * 2;
  _Float16* Oh     = (_Float16*)w; w += (size_t)BN * DIM * 2;

  k_cvt_f16<<<(BN * DIM + 255) / 256, 256, 0, stream>>>(x, xh, BN * DIM);
  k_transpose_cvt<<<(3 * DIM * DIM + 255) / 256, 256, 0, stream>>>(W_qkv, Wqkvt,
                                                                   DIM, 3 * DIM);
  k_transpose_cvt<<<(DIM * DIM + 255) / 256, 256, 0, stream>>>(W_proj, Wprojt, DIM,
                                                               DIM);
  k_qkv<<<dim3(3 * DIM / 64, BN / 64), 32, 0, stream>>>(xh, Wqkvt, b_qkv, Qh, Kh,
                                                        Vth);
  k_attn<<<BATCH * NHEAD * (SEQ / 64), 128, 0, stream>>>(Qh, Kh, Vth, Oh);
  k_proj<<<dim3(DIM / 64, BN / 64), 32, 0, stream>>>(Oh, Wprojt, b_proj,
                                                     (float*)d_out);
}
